// MultiPatternRNN_63410897158756
// MI455X (gfx1250) — matrix operand order repaired and checked by reading, not compiled, not run on hardware
//
#include <hip/hip_runtime.h>
#include <hip/hip_bf16.h>

typedef __attribute__((ext_vector_type(16))) _Float16 v16h;
typedef __attribute__((ext_vector_type(8)))  _Float16 v8h;
typedef __attribute__((ext_vector_type(8)))  float    v8f;

#define BATCH 256
#define SEQ   4096
#define HID   64
#define G3    192      // 3*H, gate order r,z,n
#define BT    16       // batch tile per workgroup
#define NTHREADS 256
#define NWAVES 8

// ---------------- LDS layout for the fused scan kernel (dynamic LDS) ----------------
#define SMEM_BYTES (122880 /*5 f16 weight mats*/ + 5376 /*wih0+6 biases f32*/ \
                    + 24576 /*hh+xi gate bufs*/ + 12288 /*h32 x3*/ + 6144 /*h16 x3*/ + 64 /*xs*/)

// ---------------- fast transcendental helpers ----------------
__device__ __forceinline__ float fast_rcp(float x) {
#if __has_builtin(__builtin_amdgcn_rcpf)
    return __builtin_amdgcn_rcpf(x);
#else
    return 1.f / x;
#endif
}
__device__ __forceinline__ float fast_sigmoid(float v) {
    return fast_rcp(1.f + __expf(-v));
}
__device__ __forceinline__ float fast_tanh(float v) {
#if __has_builtin(__builtin_amdgcn_tanhf)
    return __builtin_amdgcn_tanhf(v);          // gfx1250 v_tanh_f32
#else
    return 1.f - 2.f * fast_rcp(1.f + __expf(2.f * v));
#endif
}

// A fragment (16x32 f16, MxK) from LDS row-major [16][64] hidden state: lane holds row (lane&15), contiguous K run.
__device__ __forceinline__ v16h load_a_frag(const _Float16* h16, int kc, int lane) {
    const _Float16* p = h16 + (lane & 15) * HID + kc * 32 + (lane >> 4) * 16;
    v8h lo = *(const v8h*)(p);
    v8h hi = *(const v8h*)(p + 8);
    v16h r;
#pragma unroll
    for (int i = 0; i < 8; ++i) { r[i] = lo[i]; r[i + 8] = hi[i]; }
    return r;
}

// B fragment (32x16 f16, KxN): lane holds column (lane&15), contiguous K run.
// A row of the row-major [192][64] weight is a column of B.
__device__ __forceinline__ v16h load_b_frag(const _Float16* w, int n0, int kc, int lane) {
    const _Float16* p = w + (n0 * 16 + (lane & 15)) * HID + kc * 32 + (lane >> 4) * 16;
    return *(const v16h*)p;
}

__device__ __forceinline__ v8f wmma_f16(v16h a, v16h b, v8f c) {
    return __builtin_amdgcn_wmma_f32_16x16x32_f16(false, a, false, b, (short)0, c, false, false);
}

// C/D layout: VGPR i -> M = i + (lane>=16 ? 8 : 0), N = lane&15.
__device__ __forceinline__ void store_c(v8f c, float* dst, int n0, int lane) {
    int n  = n0 * 16 + (lane & 15);
    int mb = (lane >> 4) * 8;
#pragma unroll
    for (int i = 0; i < 8; ++i) dst[(mb + i) * G3 + n] = c[i];
}

// GRU gate math. MODE 0: layer0 (xi = xs*wih0+bih0 inline). MODE 1: mid. MODE 2: final (+store).
template <int MODE>
__device__ __forceinline__ void gate_update(int tid, const float* s_xi, const float* s_hh,
                                            const float* s_wih0, const float* s_bih0,
                                            const float* s_xs,
                                            const float* bih, const float* bhh,
                                            float* h32, _Float16* h16,
                                            _Float16* out, int b0, int s) {
    int row = tid >> 4;
    int c0  = (tid & 15) * 4;
    float xv = (MODE == 0) ? s_xs[row] : 0.f;
#pragma unroll
    for (int k = 0; k < 4; ++k) {
        int c = c0 + k;
        float xr, xz, xn;
        if (MODE == 0) {
            xr = xv * s_wih0[c]       + s_bih0[c];
            xz = xv * s_wih0[64 + c]  + s_bih0[64 + c];
            xn = xv * s_wih0[128 + c] + s_bih0[128 + c];
        } else {
            xr = s_xi[row * G3 + c]       + bih[c];
            xz = s_xi[row * G3 + 64 + c]  + bih[64 + c];
            xn = s_xi[row * G3 + 128 + c] + bih[128 + c];
        }
        float hr = s_hh[row * G3 + c]       + bhh[c];
        float hz = s_hh[row * G3 + 64 + c]  + bhh[64 + c];
        float hn = s_hh[row * G3 + 128 + c] + bhh[128 + c];
        float r = fast_sigmoid(xr + hr);
        float z = fast_sigmoid(xz + hz);
        float n = fast_tanh(xn + r * hn);
        float h = (1.f - z) * n + z * h32[row * HID + c];
        h32[row * HID + c] = h;
        h16[row * HID + c] = (_Float16)h;
        if (MODE == 2) out[((size_t)(b0 + row) * SEQ + s) * HID + c] = (_Float16)h;
    }
}

// ---------------- Fused 3-layer GRU scan: one WG = 16 batch rows, full sequence ----------------
__global__ __launch_bounds__(NTHREADS)
void gru3_scan_kernel(const float* __restrict__ x,
                      const float* __restrict__ wih0, const float* __restrict__ whh0,
                      const float* __restrict__ bih0, const float* __restrict__ bhh0,
                      const float* __restrict__ wih1, const float* __restrict__ whh1,
                      const float* __restrict__ bih1, const float* __restrict__ bhh1,
                      const float* __restrict__ wih2, const float* __restrict__ whh2,
                      const float* __restrict__ bih2, const float* __restrict__ bhh2,
                      _Float16* __restrict__ out) {
    extern __shared__ char smem[];
    _Float16* s_whh0 = (_Float16*)smem;
    _Float16* s_whh1 = s_whh0 + G3 * HID;
    _Float16* s_whh2 = s_whh1 + G3 * HID;
    _Float16* s_wih1 = s_whh2 + G3 * HID;
    _Float16* s_wih2 = s_wih1 + G3 * HID;
    float* s_wih0 = (float*)(s_wih2 + G3 * HID);
    float* s_bih0 = s_wih0 + G3;
    float* s_bih1 = s_bih0 + G3;
    float* s_bih2 = s_bih1 + G3;
    float* s_bhh0 = s_bih2 + G3;
    float* s_bhh1 = s_bhh0 + G3;
    float* s_bhh2 = s_bhh1 + G3;
    float* s_hh   = s_bhh2 + G3;                          // [16][192]
    float* s_xi   = s_hh + BT * G3;                       // [16][192]
    float* s_h32  = s_xi + BT * G3;                       // 3 x [16][64]
    _Float16* s_h16 = (_Float16*)(s_h32 + 3 * BT * HID);  // 3 x [16][64]
    float* s_xs   = (float*)(s_h16 + 3 * BT * HID);       // [16]

    float*    h32_0 = s_h32;  float*    h32_1 = s_h32 + BT * HID;  float*    h32_2 = s_h32 + 2 * BT * HID;
    _Float16* h16_0 = s_h16;  _Float16* h16_1 = s_h16 + BT * HID;  _Float16* h16_2 = s_h16 + 2 * BT * HID;

    const int tid  = threadIdx.x;
    const int lane = tid & 31;
    const int wave = __builtin_amdgcn_readfirstlane(tid >> 5);   // scalar -> s_cbranch, EXEC untouched
    const int b0   = blockIdx.x * BT;

    // ---- stage weights into LDS (f32 -> f16) ----
    for (int i = tid; i < G3 * HID; i += NTHREADS) {
        s_whh0[i] = (_Float16)whh0[i];
        s_whh1[i] = (_Float16)whh1[i];
        s_whh2[i] = (_Float16)whh2[i];
        s_wih1[i] = (_Float16)wih1[i];
        s_wih2[i] = (_Float16)wih2[i];
    }
    for (int i = tid; i < G3; i += NTHREADS) {
        s_wih0[i] = wih0[i];
        s_bih0[i] = bih0[i]; s_bih1[i] = bih1[i]; s_bih2[i] = bih2[i];
        s_bhh0[i] = bhh0[i]; s_bhh1[i] = bhh1[i]; s_bhh2[i] = bhh2[i];
    }
    for (int i = tid; i < 3 * BT * HID; i += NTHREADS) { s_h32[i] = 0.f; s_h16[i] = (_Float16)0.f; }
    if (tid < BT) s_xs[tid] = x[(size_t)(b0 + tid) * SEQ];
    __syncthreads();

    // ---- preload step-invariant B fragments into registers ----
    // layer 0: 12 HH tiles -> waves 0..7 take tile `wave`, waves 0..3 also take `wave+8`.
    // layers 1/2: 24 tiles (12 HH + 12 XI) -> 3 tiles per wave: t = wave + 8*j.
    v16h B0a0 = load_b_frag(s_whh0, wave, 0, lane);
    v16h B0a1 = load_b_frag(s_whh0, wave, 1, lane);
    v16h B0b0 = B0a0, B0b1 = B0a1;
    if (wave < 4) {
        B0b0 = load_b_frag(s_whh0, wave + 8, 0, lane);
        B0b1 = load_b_frag(s_whh0, wave + 8, 1, lane);
    }
    v16h B1f[3][2], B2f[3][2];
#pragma unroll
    for (int j = 0; j < 3; ++j) {
        int t = wave + 8 * j;
        int n0 = (t < 12) ? t : t - 12;
        const _Float16* w1 = (t < 12) ? s_whh1 : s_wih1;
        const _Float16* w2 = (t < 12) ? s_whh2 : s_wih2;
        B1f[j][0] = load_b_frag(w1, n0, 0, lane);
        B1f[j][1] = load_b_frag(w1, n0, 1, lane);
        B2f[j][0] = load_b_frag(w2, n0, 0, lane);
        B2f[j][1] = load_b_frag(w2, n0, 1, lane);
    }

    for (int s = 0; s < SEQ; ++s) {
        // -------- layer 0: HH WMMA (xi handled inline in the gate phase) --------
        {
            v16h a0 = load_a_frag(h16_0, 0, lane);
            v16h a1 = load_a_frag(h16_0, 1, lane);
            v8f c = {};
            c = wmma_f16(a0, B0a0, c);
            c = wmma_f16(a1, B0a1, c);
            store_c(c, s_hh, wave, lane);
            if (wave < 4) {
                v8f c2 = {};
                c2 = wmma_f16(a0, B0b0, c2);
                c2 = wmma_f16(a1, B0b1, c2);
                store_c(c2, s_hh, wave + 8, lane);
            }
        }
        __syncthreads();
        gate_update<0>(tid, s_xi, s_hh, s_wih0, s_bih0, s_xs, nullptr, s_bhh0, h32_0, h16_0, nullptr, b0, s);
        __syncthreads();

        // -------- layer 1: HH + XI WMMA (3 fixed tiles per wave) --------
        {
            v16h ah0 = load_a_frag(h16_1, 0, lane), ah1 = load_a_frag(h16_1, 1, lane);
            v16h ax0 = load_a_frag(h16_0, 0, lane), ax1 = load_a_frag(h16_0, 1, lane);
#pragma unroll
            for (int j = 0; j < 3; ++j) {
                int t = wave + 8 * j;
                bool hh = (t < 12);
                int n0 = hh ? t : t - 12;
                float* dst = hh ? s_hh : s_xi;
                v8f c = {};
                if (hh) { c = wmma_f16(ah0, B1f[j][0], c); c = wmma_f16(ah1, B1f[j][1], c); }
                else    { c = wmma_f16(ax0, B1f[j][0], c); c = wmma_f16(ax1, B1f[j][1], c); }
                store_c(c, dst, n0, lane);
            }
        }
        __syncthreads();
        gate_update<1>(tid, s_xi, s_hh, nullptr, nullptr, nullptr, s_bih1, s_bhh1, h32_1, h16_1, nullptr, b0, s);
        __syncthreads();

        // -------- layer 2: HH + XI WMMA --------
        {
            v16h ah0 = load_a_frag(h16_2, 0, lane), ah1 = load_a_frag(h16_2, 1, lane);
            v16h ax0 = load_a_frag(h16_1, 0, lane), ax1 = load_a_frag(h16_1, 1, lane);
#pragma unroll
            for (int j = 0; j < 3; ++j) {
                int t = wave + 8 * j;
                bool hh = (t < 12);
                int n0 = hh ? t : t - 12;
                float* dst = hh ? s_hh : s_xi;
                v8f c = {};
                if (hh) { c = wmma_f16(ah0, B2f[j][0], c); c = wmma_f16(ah1, B2f[j][1], c); }
                else    { c = wmma_f16(ax0, B2f[j][0], c); c = wmma_f16(ax1, B2f[j][1], c); }
                store_c(c, dst, n0, lane);
            }
        }
        __syncthreads();
        gate_update<2>(tid, s_xi, s_hh, nullptr, nullptr, nullptr, s_bih2, s_bhh2, h32_2, h16_2, out, b0, s);
        if (tid < BT && s + 1 < SEQ) s_xs[tid] = x[(size_t)(b0 + tid) * SEQ + s + 1];
        __syncthreads();
    }
}

// ---------------- Attention: one WG per batch row, online softmax over S ----------------
// K/V projections folded: scores = out . (qh @ Wk); ctx = (sum_s p_s out_s) @ Wv^T.
__global__ __launch_bounds__(256)
void attn_kernel(const _Float16* __restrict__ outseq,   // [B,S,64] f16
                 const float* __restrict__ in_proj_w, const float* __restrict__ in_proj_b,
                 const float* __restrict__ out_proj_w, const float* __restrict__ out_proj_b,
                 const float* __restrict__ fc_w, const float* __restrict__ fc_b,
                 float* __restrict__ y) {
    const int b   = blockIdx.x;
    const int tid = threadIdx.x;

    __shared__ float    sq[HID];
    __shared__ float    sqt[4][HID];
    __shared__ float    ssb[4];
    __shared__ _Float16 schunk[64][HID];
    __shared__ float    ssc[4][64];
    __shared__ float    sp[4][64];
    __shared__ float    smx[4], slm[4], sscale[4];
    __shared__ float    sacc[4][HID];
    __shared__ float    sctx[HID];
    __shared__ float    sao[HID];

    const _Float16* ob = outseq + (size_t)b * SEQ * HID;

    if (tid < HID) {   // q = out[:, S-1, :] @ Wq^T + bq
        const _Float16* last = ob + (size_t)(SEQ - 1) * HID;
        float acc = in_proj_b[tid];
        for (int k = 0; k < HID; ++k) acc += (float)last[k] * in_proj_w[tid * HID + k];
        sq[tid] = acc;
    }
    if (tid < 4) { smx[tid] = -1e30f; slm[tid] = 0.f; }
    __syncthreads();

    const int h = tid >> 6, e = tid & 63;
    {   // qtilde[h][e] = sum_d q[h*16+d] * Wk[h*16+d][e]
        float acc = 0.f;
        for (int d = 0; d < 16; ++d) acc += sq[h * 16 + d] * in_proj_w[(64 + h * 16 + d) * HID + e];
        sqt[h][e] = acc;
    }
    if (tid < 4) {
        float acc = 0.f;
        for (int d = 0; d < 16; ++d) acc += sq[tid * 16 + d] * in_proj_b[64 + tid * 16 + d];
        ssb[tid] = acc;
    }
    float accv = 0.f;
    __syncthreads();

    for (int s0 = 0; s0 < SEQ; s0 += 64) {
        const _Float16* src = ob + (size_t)s0 * HID;
        {
            const v8h* vs = (const v8h*)src;
            v8h* vd = (v8h*)&schunk[0][0];
            for (int i = tid; i < 64 * HID / 8; i += 256) vd[i] = vs[i];
        }
        if (s0 + 64 < SEQ) __builtin_prefetch(src + 64 * HID, 0, 1);
        __syncthreads();
        {   // score[h][s=e]
            float acc = 0.f;
            for (int k = 0; k < HID; ++k) acc += (float)schunk[e][k] * sqt[h][k];
            ssc[h][e] = (acc + ssb[h]) * 0.25f;   // 1/sqrt(HD)
        }
        __syncthreads();
        if (tid < 4) {
            float mx = smx[tid];
            for (int i = 0; i < 64; ++i) mx = fmaxf(mx, ssc[tid][i]);
            float sum = 0.f;
            for (int i = 0; i < 64; ++i) { float p = __expf(ssc[tid][i] - mx); sp[tid][i] = p; sum += p; }
            float sc = __expf(smx[tid] - mx);
            slm[tid] = slm[tid] * sc + sum;
            smx[tid] = mx;
            sscale[tid] = sc;
        }
        __syncthreads();
        {
            float a = accv * sscale[h];
            for (int i = 0; i < 64; ++i) a += sp[h][i] * (float)schunk[i][e];
            accv = a;
        }
        __syncthreads();
    }
    sacc[h][e] = accv * fast_rcp(slm[h]);
    __syncthreads();
    if (tid < HID) {   // ctx = u @ Wv^T + bv
        int hh = tid >> 4, d = tid & 15;
        float acc = in_proj_b[128 + hh * 16 + d];
        for (int k = 0; k < HID; ++k) acc += sacc[hh][k] * in_proj_w[(128 + hh * 16 + d) * HID + k];
        sctx[tid] = acc;
    }
    __syncthreads();
    if (tid < HID) {   // attn_out = ctx @ out_proj_w^T + out_proj_b
        float acc = out_proj_b[tid];
        for (int i = 0; i < HID; ++i) acc += sctx[i] * out_proj_w[tid * HID + i];
        sao[tid] = acc;
    }
    __syncthreads();
    if (tid == 0) {
        float acc = fc_b[0];
        for (int j = 0; j < HID; ++j) acc += sao[j] * fc_w[j];
        y[b] = fast_sigmoid(acc);
    }
}

extern "C" void kernel_launch(void* const* d_in, const int* in_sizes, int n_in,
                              void* d_out, int out_size, void* d_ws, size_t ws_size,
                              hipStream_t stream) {
    const float* x     = (const float*)d_in[0];
    const float* wih0  = (const float*)d_in[1];
    const float* whh0  = (const float*)d_in[2];
    const float* bih0  = (const float*)d_in[3];
    const float* bhh0  = (const float*)d_in[4];
    const float* wih1  = (const float*)d_in[5];
    const float* whh1  = (const float*)d_in[6];
    const float* bih1  = (const float*)d_in[7];
    const float* bhh1  = (const float*)d_in[8];
    const float* wih2  = (const float*)d_in[9];
    const float* whh2  = (const float*)d_in[10];
    const float* bih2  = (const float*)d_in[11];
    const float* bhh2  = (const float*)d_in[12];
    const float* ipw   = (const float*)d_in[13];
    const float* ipb   = (const float*)d_in[14];
    const float* opw   = (const float*)d_in[15];
    const float* opb   = (const float*)d_in[16];
    const float* fcw   = (const float*)d_in[17];
    const float* fcb   = (const float*)d_in[18];

    _Float16* outseq = (_Float16*)d_ws;   // [B,S,64] f16 final-layer activations
    float* y = (float*)d_out;

    gru3_scan_kernel<<<BATCH / BT, NTHREADS, SMEM_BYTES, stream>>>(
        x, wih0, whh0, bih0, bhh0, wih1, whh1, bih1, bhh1, wih2, whh2, bih2, bhh2, outseq);
    attn_kernel<<<BATCH, 256, 0, stream>>>(outseq, ipw, ipb, opw, opb, fcw, fcb, y);
}
